// EncoderLayer_23081154249487
// MI455X (gfx1250) — matrix operand order repaired and checked by reading, not compiled, not run on hardware
//
#include <hip/hip_runtime.h>
#include <hip/hip_bf16.h>
#include <math.h>
#include <stdint.h>

typedef __attribute__((ext_vector_type(16))) __bf16       v16bf;
typedef __attribute__((ext_vector_type(8)))  float        v8f;
typedef __attribute__((ext_vector_type(4)))  unsigned int v4u;
typedef __attribute__((ext_vector_type(8)))  int          v8i;
typedef __attribute__((ext_vector_type(4)))  int          v4i;

#define B_   4096
#define D_   256
#define H_   8
#define L_   64
#define HD_  32
#define DFF_ 1024

// ---- WMMA fragment loaders (bf16, 16x16x32) -------------------------------
// A: 16x32 tile, row-major bf16 with element stride `stride`.
// lanes 0-15: row=lane, K = 0..15 contiguous
// lanes 16-31: row=lane-16, K = 16..31 contiguous
__device__ __forceinline__ v16bf ld_afrag(const __bf16* tile, int stride, int lane) {
  int row = lane & 15;
  int kb  = (lane >> 4) << 4;
  const __bf16* p = tile + row * stride + kb;
  v16bf a;
#pragma unroll
  for (int e = 0; e < 8; ++e) a[e] = p[e];
#pragma unroll
  for (int e = 0; e < 8; ++e) a[e + 8] = p[e + 8];
  return a;
}

// B: 32x16 tile from a column-major (transposed) weight WT[N][K]:
// lane n (0-15): col=n, K = 0..15 contiguous; lanes 16-31: col=n-16, K = 16..31.
__device__ __forceinline__ v16bf ld_bfrag(const __bf16* tileT, int strideK, int lane) {
  int col = lane & 15;
  int kb  = (lane >> 4) << 4;
  const __bf16* p = tileT + col * strideK + kb;
  v16bf b;
#pragma unroll
  for (int e = 0; e < 16; ++e) b[e] = p[e];
  return b;
}

__device__ __forceinline__ v8f wmma_bf16(v16bf a, v16bf b, v8f c) {
  return __builtin_amdgcn_wmma_f32_16x16x32_bf16(false, a, false, b, (short)0, c, false, false);
}

// ---- TDM: async-load a 64x256 bf16 tile (row stride = rowStrideElems) -----
// into LDS with 16B padding every 512B -> padded LDS stride of 264 elements.
// D# layout per CDNA5 ISA. Issued by one wave,
// completion via TENSORcnt. This toolchain exposes the 6-arg builtin:
// (uint32x4 g0, int32x8 g1, int32x4 g2, int32x4 g3, int32x8, i32 cpol).
__device__ __forceinline__ void tdm_load_tile_64x256(const __bf16* gsrc,
                                                     int rowStrideElems,
                                                     __bf16* lds_dst) {
#if __has_builtin(__builtin_amdgcn_tensor_load_to_lds)
  unsigned long long ga     = (unsigned long long)(uintptr_t)gsrc;
  unsigned int       ldsoff = (unsigned int)(uintptr_t)lds_dst;  // low 32b = LDS offset
  v4u g0;
  g0[0] = 1u;                                    // count=1, is_restore=0, gather=0
  g0[1] = ldsoff;                                // lds_addr (bytes)
  g0[2] = (unsigned int)ga;                      // global_addr[31:0]
  g0[3] = (unsigned int)(ga >> 32) | (2u << 30); // global_addr[56:32] | type=2
  v8i g1;
  // data_size=1(2B) | pad_enable | pad_interval=6(128 DW) | pad_amount=3(4 DW)
  g1[0] = (int)((1u << 16) | (1u << 20) | (6u << 22) | (3u << 25));
  g1[1] = (int)(256u << 16);        // tensor_dim0 = 256 (bits 79:48, low half)
  g1[2] = (int)(64u << 16);         // tensor_dim0 hi=0 | tensor_dim1 = 64
  g1[3] = (int)(256u << 16);        // tensor_dim1 hi=0 | tile_dim0 = 256
  g1[4] = 64;                       // tile_dim1 = 64 | tile_dim2 = 0
  g1[5] = rowStrideElems;           // tensor_dim0_stride (elements)
  g1[6] = 0;
  g1[7] = 0;
  v4i z4 = {0, 0, 0, 0};
  v8i z8 = {0, 0, 0, 0, 0, 0, 0, 0};
  __builtin_amdgcn_tensor_load_to_lds(g0, g1, z4, z4, z8, 0);
  __builtin_amdgcn_s_wait_tensorcnt((short)0);
#else
  (void)gsrc; (void)rowStrideElems; (void)lds_dst;
#endif
}

#if __has_builtin(__builtin_amdgcn_tensor_load_to_lds)
#define HAVE_TDM 1
#else
#define HAVE_TDM 0
#endif

// Stage a 64x256 bf16 tile from global A (row stride strideElems) into Alds
// (padded stride 264). Caller must have synchronized before; synchronizes after.
__device__ __forceinline__ void stage_tile(const __bf16* __restrict__ A,
                                           int strideElems, __bf16* Alds,
                                           int tid) {
#if HAVE_TDM
  if (tid < 32) tdm_load_tile_64x256(A, strideElems, Alds);
#else
  for (int i = tid; i < 64 * 256; i += 256) {
    int r = i >> 8, c = i & 255;
    Alds[r * 264 + c] = A[(size_t)r * strideElems + c];
  }
#endif
  __syncthreads();
}

// ---- Kernel: convert f32 weight [K,N] -> bf16 transposed [N,K] ------------
__global__ void wconv_t(const float* __restrict__ in, __bf16* __restrict__ out,
                        int K, int N) {
  int total = K * N;
  for (int i = blockIdx.x * blockDim.x + threadIdx.x; i < total;
       i += gridDim.x * blockDim.x) {
    int k = i / N;
    int n = i - k * N;
    out[n * K + k] = (__bf16)in[i];
  }
}

// ---- Kernel: gather query-node rows (f32 + bf16 copies) -------------------
__global__ __launch_bounds__(256)
void gather_x(const int* __restrict__ batch, const float* __restrict__ emb,
              float* __restrict__ xF, __bf16* __restrict__ xB) {
  int i = blockIdx.x * blockDim.x + threadIdx.x;  // exactly B_*D_ threads
  int b = i >> 8, d = i & 255;
  float v = emb[(size_t)batch[b] * D_ + d];
  xF[i] = v;
  xB[i] = (__bf16)v;
}

// ---- Kernel: C = act((A @ W^T + bias) * scale) ----------------------------
// A: [B_,256] bf16; WT: [NTOT,256] bf16; 64 rows per block.
// Each wave owns NTOT/128 nt-columns and all 4 mt rows -> every B fragment is
// loaded exactly once per block.
template <int NTOT, bool RELU>
__global__ __launch_bounds__(256)
void gemm_act(const __bf16* __restrict__ A, const __bf16* __restrict__ WT,
              const float* __restrict__ bias, float scale,
              float* __restrict__ outF, __bf16* __restrict__ outB) {
  const int AS = 264;
  __shared__ __bf16 Alds[64 * AS];
  const int tid = threadIdx.x, lane = tid & 31, w = tid >> 5;
  const int b0  = blockIdx.x * 64;

  stage_tile(A + (size_t)b0 * D_, D_, Alds, tid);

  const int NT  = NTOT / 16;  // col tiles
  const int NPW = NT / 8;     // nt per wave
  const v8f zero = {0.f, 0.f, 0.f, 0.f, 0.f, 0.f, 0.f, 0.f};
  for (int j = 0; j < NPW; ++j) {
    int nt = w * NPW + j;
    v8f acc[4];
#pragma unroll
    for (int mt = 0; mt < 4; ++mt) acc[mt] = zero;
#pragma unroll
    for (int kk = 0; kk < 8; ++kk) {
      v16bf bb = ld_bfrag(WT + (nt * 16) * 256 + kk * 32, 256, lane);
#pragma unroll
      for (int mt = 0; mt < 4; ++mt) {
        v16bf a = ld_afrag(&Alds[(mt * 16) * AS + kk * 32], AS, lane);
        acc[mt] = wmma_bf16(a, bb, acc[mt]);
      }
    }
    int   col = nt * 16 + (lane & 15);
    float bi  = bias[col];
#pragma unroll
    for (int mt = 0; mt < 4; ++mt) {
      int rbase = mt * 16 + ((lane >> 4) << 3);
#pragma unroll
      for (int e = 0; e < 8; ++e) {
        float v = (acc[mt][e] + bi) * scale;
        if (RELU) v = v > 0.f ? v : 0.f;
        size_t idx = (size_t)(b0 + rbase + e) * NTOT + col;
        if (outF) outF[idx] = v;
        if (outB) outB[idx] = (__bf16)v;
      }
    }
  }
}

// ---- Kernel: out = LN(resid + act(A @ W^T + bias)) ------------------------
// A: [B_,KTOT] bf16; WT: [256,KTOT] bf16; 64 rows per block; N fixed 256.
template <int KTOT, bool RELU>
__global__ __launch_bounds__(256)
void gemm_res_ln(const __bf16* __restrict__ A, const __bf16* __restrict__ WT,
                 const float* __restrict__ bias, const float* __restrict__ resid,
                 const float* __restrict__ lnw, const float* __restrict__ lnb,
                 float* __restrict__ outF, __bf16* __restrict__ outB) {
  const int AS = 264;
  const int FS = 260;
  __shared__ __bf16 Alds[64 * AS];
  __shared__ float  flds[64 * FS];
  const int tid = threadIdx.x, lane = tid & 31, w = tid >> 5;
  const int b0  = blockIdx.x * 64;

  const v8f zero = {0.f, 0.f, 0.f, 0.f, 0.f, 0.f, 0.f, 0.f};
  v8f acc[2][4];  // [nt-per-wave][mt]
#pragma unroll
  for (int j = 0; j < 2; ++j)
#pragma unroll
    for (int mt = 0; mt < 4; ++mt) acc[j][mt] = zero;

  const int CH = KTOT / 256;
  for (int ch = 0; ch < CH; ++ch) {
    const int K0 = ch * 256;
    __syncthreads();  // previous Alds fully consumed
    stage_tile(A + (size_t)b0 * KTOT + K0, KTOT, Alds, tid);
    for (int j = 0; j < 2; ++j) {
      int nt = w * 2 + j;
#pragma unroll
      for (int kk = 0; kk < 8; ++kk) {
        v16bf bb = ld_bfrag(WT + (nt * 16) * KTOT + K0 + kk * 32, KTOT, lane);
#pragma unroll
        for (int mt = 0; mt < 4; ++mt) {
          v16bf a = ld_afrag(&Alds[(mt * 16) * AS + kk * 32], AS, lane);
          acc[j][mt] = wmma_bf16(a, bb, acc[j][mt]);
        }
      }
    }
  }

  for (int j = 0; j < 2; ++j) {
    int   nt  = w * 2 + j;
    int   col = nt * 16 + (lane & 15);
    float bi  = bias[col];
#pragma unroll
    for (int mt = 0; mt < 4; ++mt) {
      int rbase = mt * 16 + ((lane >> 4) << 3);
#pragma unroll
      for (int e = 0; e < 8; ++e) {
        float v = acc[j][mt][e] + bi;
        if (RELU) v = v > 0.f ? v : 0.f;
        flds[(rbase + e) * FS + col] = v;
      }
    }
  }
  __syncthreads();

  // residual + layernorm: wave w handles rows w*8..w*8+7, lane covers 8 cols
  for (int j = 0; j < 8; ++j) {
    int   row = w * 8 + j;
    float y[8];
    float s = 0.f, s2 = 0.f;
#pragma unroll
    for (int q = 0; q < 8; ++q) {
      int   c = lane + q * 32;
      float v = resid[(size_t)(b0 + row) * D_ + c] + flds[row * FS + c];
      y[q] = v;
      s += v;
      s2 += v * v;
    }
    for (int off = 16; off; off >>= 1) {
      s  += __shfl_xor(s, off, 32);
      s2 += __shfl_xor(s2, off, 32);
    }
    float mu   = s * (1.f / 256.f);
    float var  = s2 * (1.f / 256.f) - mu * mu;
    float rstd = rsqrtf(var + 1e-5f);
#pragma unroll
    for (int q = 0; q < 8; ++q) {
      int    c   = lane + q * 32;
      float  v   = (y[q] - mu) * rstd * lnw[c] + lnb[c];
      size_t idx = (size_t)(b0 + row) * D_ + c;
      outF[idx] = v;
      if (outB) outB[idx] = (__bf16)v;
    }
  }
}

// ---- Kernel: gather + K/V WMMA + masked softmax + attout ------------------
__global__ __launch_bounds__(256)
void attn_kernel(const int* __restrict__ nei_idx,
                 const unsigned char* __restrict__ nei_mask,
                 const float* __restrict__ emb, const float* __restrict__ qF,
                 const __bf16* __restrict__ WkT, const float* __restrict__ bk,
                 const __bf16* __restrict__ WvT, const float* __restrict__ bv,
                 __bf16* __restrict__ aoB) {
  const int AS = 264;
  __shared__ __bf16 neiA[64 * AS];
  __shared__ __bf16 Klds[64 * AS];
  __shared__ __bf16 Vlds[64 * AS];
  __shared__ float  ql[256];
  __shared__ float  att[8 * 64];

  const int tid  = threadIdx.x;
  const int lane = tid & 31;
  const int w    = tid >> 5;
  const int b    = blockIdx.x;

  // pre-scaled q row (from WMMA q-projection kernel)
  ql[tid] = qF[(size_t)b * D_ + tid];

  // prefetch neighbor row bases (global_prefetch_b8) before the gather loop
  if (tid < 64)
    __builtin_prefetch(&emb[(size_t)nei_idx[b * L_ + tid] * D_], 0, 0);

  // gather neighbors -> bf16 LDS (embedding table is L2-resident: 102MB < 192MB)
  for (int i = tid; i < 64 * 256; i += 256) {
    int l = i >> 8, c = i & 255;
    neiA[l * AS + c] = (__bf16)emb[(size_t)nei_idx[b * L_ + l] * D_ + c];
  }
  __syncthreads();

  // K = nei@Wk + bk, V = nei@Wv + bv via WMMA.
  // Each wave owns 2 nt-columns and all 4 mt rows with 4+4 live accumulators:
  // every Wk/Wv B-fragment is fetched exactly once per block (4x less L2
  // traffic than mt-per-wave assignment; A re-reads hit LDS).
  const v8f zero = {0.f, 0.f, 0.f, 0.f, 0.f, 0.f, 0.f, 0.f};
  for (int j = 0; j < 2; ++j) {
    int nt = w * 2 + j;
    v8f ck[4], cv[4];
#pragma unroll
    for (int mt = 0; mt < 4; ++mt) { ck[mt] = zero; cv[mt] = zero; }
#pragma unroll
    for (int kk = 0; kk < 8; ++kk) {
      v16bf wkf = ld_bfrag(WkT + (nt * 16) * 256 + kk * 32, 256, lane);
      v16bf wvf = ld_bfrag(WvT + (nt * 16) * 256 + kk * 32, 256, lane);
#pragma unroll
      for (int mt = 0; mt < 4; ++mt) {
        v16bf a = ld_afrag(&neiA[(mt * 16) * AS + kk * 32], AS, lane);
        ck[mt] = wmma_bf16(a, wkf, ck[mt]);
        cv[mt] = wmma_bf16(a, wvf, cv[mt]);
      }
    }
    int   col = nt * 16 + (lane & 15);
    float bkc = bk[col], bvc = bv[col];
#pragma unroll
    for (int mt = 0; mt < 4; ++mt) {
      int rbase = mt * 16 + ((lane >> 4) << 3);
#pragma unroll
      for (int e = 0; e < 8; ++e) {
        Klds[(rbase + e) * AS + col] = (__bf16)(ck[mt][e] + bkc);
        Vlds[(rbase + e) * AS + col] = (__bf16)(cv[mt][e] + bvc);
      }
    }
  }
  __syncthreads();

  // scores + masked softmax; wave w == head h, each lane does l and l+32
  {
    const int h = w;
    const unsigned char* mrow = nei_mask + (size_t)b * L_;
    const int l0 = lane, l1 = lane + 32;
    float s0 = 0.f, s1 = 0.f;
    for (int j = 0; j < HD_; ++j) {
      float qv = ql[h * HD_ + j];
      s0 += qv * (float)Klds[l0 * AS + h * HD_ + j];
      s1 += qv * (float)Klds[l1 * AS + h * HD_ + j];
    }
    if (mrow[l0]) s0 = -INFINITY;
    if (mrow[l1]) s1 = -INFINITY;
    float m = fmaxf(s0, s1);
    for (int off = 16; off; off >>= 1) m = fmaxf(m, __shfl_xor(m, off, 32));
    float e0 = __expf(s0 - m), e1 = __expf(s1 - m);
    float se = e0 + e1;
    for (int off = 16; off; off >>= 1) se += __shfl_xor(se, off, 32);
    float inv = 1.f / se;
    att[h * 64 + l0] = e0 * inv;
    att[h * 64 + l1] = e1 * inv;
  }
  __syncthreads();

  // attout[d] = sum_l att[h(d)][l] * V[l][d] -> bf16 for the Wo WMMA GEMM
  {
    const int h   = tid >> 5;
    float     acc = 0.f;
    for (int l = 0; l < 64; ++l) acc += att[h * 64 + l] * (float)Vlds[l * AS + tid];
    aoB[(size_t)b * D_ + tid] = (__bf16)acc;
  }
}

// ---- host launch ----------------------------------------------------------
extern "C" void kernel_launch(void* const* d_in, const int* in_sizes, int n_in,
                              void* d_out, int out_size, void* d_ws, size_t ws_size,
                              hipStream_t stream) {
  (void)in_sizes; (void)n_in; (void)out_size; (void)ws_size;

  const int*           batch    = (const int*)d_in[0];
  const int*           nei_idx  = (const int*)d_in[1];
  const unsigned char* nei_mask = (const unsigned char*)d_in[2];
  const float*         emb      = (const float*)d_in[3];
  const float* Wq = (const float*)d_in[4];  const float* bq = (const float*)d_in[5];
  const float* Wk = (const float*)d_in[6];  const float* bk = (const float*)d_in[7];
  const float* Wv = (const float*)d_in[8];  const float* bv = (const float*)d_in[9];
  const float* Wo = (const float*)d_in[10]; const float* bo = (const float*)d_in[11];
  const float* W1 = (const float*)d_in[12]; const float* b1 = (const float*)d_in[13];
  const float* W2 = (const float*)d_in[14]; const float* b2 = (const float*)d_in[15];
  const float* ln1w = (const float*)d_in[16]; const float* ln1b = (const float*)d_in[17];
  const float* ln2w = (const float*)d_in[18]; const float* ln2b = (const float*)d_in[19];
  float* out = (float*)d_out;

  char*   ws  = (char*)d_ws;
  __bf16* WqT = (__bf16*)(ws + 0);         //  256x256 bf16, 128 KB
  __bf16* WkT = (__bf16*)(ws + 131072);
  __bf16* WvT = (__bf16*)(ws + 262144);
  __bf16* WoT = (__bf16*)(ws + 393216);
  __bf16* W1T = (__bf16*)(ws + 524288);    // 1024x256 bf16, 512 KB
  __bf16* W2T = (__bf16*)(ws + 1048576);   //  256x1024 bf16, 512 KB
  float*  xF  = (float*)(ws + 1572864);    // B x 256 f32, 4 MB
  __bf16* xB  = (__bf16*)(ws + 5767168);   // B x 256 bf16, 2 MB
  float*  qF  = (float*)(ws + 7864320);    // B x 256 f32, 4 MB
  __bf16* aoB = (__bf16*)(ws + 12058624);  // B x 256 bf16, 2 MB
  float*  sF  = (float*)(ws + 14155776);   // B x 256 f32, 4 MB
  __bf16* sB  = (__bf16*)(ws + 18350080);  // B x 256 bf16, 2 MB
  __bf16* hB  = (__bf16*)(ws + 20447232);  // B x 1024 bf16, 8 MB

  // weights -> bf16 transposed (L2-resident, read by all GEMMs)
  wconv_t<<<64, 256, 0, stream>>>(Wq, WqT, 256, 256);
  wconv_t<<<64, 256, 0, stream>>>(Wk, WkT, 256, 256);
  wconv_t<<<64, 256, 0, stream>>>(Wv, WvT, 256, 256);
  wconv_t<<<64, 256, 0, stream>>>(Wo, WoT, 256, 256);
  wconv_t<<<256, 256, 0, stream>>>(W1, W1T, 256, 1024);
  wconv_t<<<256, 256, 0, stream>>>(W2, W2T, 1024, 256);

  // x = embeddings[batch]
  gather_x<<<(B_ * D_) / 256, 256, 0, stream>>>(batch, emb, xF, xB);

  // q = (x @ Wq + bq) * HD^-0.5   (WMMA)
  gemm_act<256, false><<<B_ / 64, 256, 0, stream>>>(
      xB, WqT, bq, 0.17677669529663687f, qF, (__bf16*)nullptr);

  // attention core: K/V projection (WMMA) + softmax + attout
  attn_kernel<<<B_, 256, 0, stream>>>(nei_idx, nei_mask, emb, qF, WkT, bk, WvT,
                                      bv, aoB);

  // s = LN1(x + ao @ Wo + bo)     (WMMA)
  gemm_res_ln<256, false><<<B_ / 64, 256, 0, stream>>>(
      aoB, WoT, bo, xF, ln1w, ln1b, sF, sB);

  // hidden = relu(s @ W1 + b1)    (WMMA)
  gemm_act<1024, true><<<B_ / 64, 256, 0, stream>>>(
      sB, W1T, b1, 1.0f, (float*)nullptr, hB);

  // out = LN2(s + relu(hidden @ W2 + b2))   (WMMA)
  gemm_res_ln<1024, true><<<B_ / 64, 256, 0, stream>>>(
      hB, W2T, b2, sF, ln2w, ln2b, out, (__bf16*)nullptr);
}
